// Encoder_41583873360305
// MI455X (gfx1250) — compile-verified, matrix operand order checked by reading, not run on hardware
//
#include <hip/hip_runtime.h>

// Problem constants (match reference)
#define BB  64
#define TT  24
#define NN  184
#define HID 128
#define G3  (3*HID)          // 384
#define RPB 64               // rows (GRU chains) per block
#define HSIZE (BB*TT*NN*HID) // H output elements

typedef __attribute__((ext_vector_type(16))) _Float16 v16h;
typedef __attribute__((ext_vector_type(8)))  _Float16 v8h;
typedef __attribute__((ext_vector_type(8)))  float    v8f;
typedef __attribute__((ext_vector_type(4)))  float    v4f;
typedef __attribute__((ext_vector_type(4)))  int      v4i;

// Fast transcendental paths: v_exp_f32 + v_rcp_f32 (and v_tanh_f32 on gfx1250).
__device__ __forceinline__ float sigf(float x) {
    return __builtin_amdgcn_rcpf(1.0f + __expf(-x));
}
__device__ __forceinline__ float tanhf_(float x) {
#if __has_builtin(__builtin_amdgcn_tanhf)
    return __builtin_amdgcn_tanhf(x);
#else
    float e = __expf(2.0f * x);
    return (e - 1.0f) * __builtin_amdgcn_rcpf(e + 1.0f);
#endif
}

// Builds a v16h A/B fragment from two contiguous 16B LDS chunks.
// Layout per CDNA5 ISA (16-bit A 16x32 / mirrored B 32x16):
//   lanes 0-15  : K = base+0..7 (v0-3), base+16..23 (v4-7)
//   lanes 16-31 : K = base+8..15,       base+24..31   (base offset hh8=8 folded in)
__device__ __forceinline__ v16h load_frag(const _Float16* p) {
    v8h lo = *(const v8h*)p;
    v8h hi = *(const v8h*)(p + 16);
    v16h f;
#pragma unroll
    for (int q = 0; q < 8; ++q) { f[q] = lo[q]; f[q+8] = hi[q]; }
    return f;
}

__global__ __launch_bounds__(256)
void gru_encoder_wmma(const float* __restrict__ pm25,   // (B,T,N,1)
                      const float* __restrict__ Wih,    // (384,2)
                      const float* __restrict__ Whh,    // (384,128)
                      const float* __restrict__ bih,    // (384)
                      const float* __restrict__ bhh,    // (384)
                      const float* __restrict__ fcw,    // (1,128)
                      const float* __restrict__ fcb,    // (1)
                      float* __restrict__ out)          // H (B,T,N,128) ++ xn (B*N)
{
    extern __shared__ __align__(16) char smem[];
    _Float16* sWhh = (_Float16*)smem;                               // 384*128 f16  (96 KB)
    _Float16* sH   = (_Float16*)(smem + G3*HID*2);                  // 2 x 64*128 f16 (32 KB)
    float*    sPm  = (float*)(smem + G3*HID*2 + 2*RPB*HID*2);       // 64 f32
    float*    sX   = sPm + RPB;                                     // 2 x 64 f32
    int*      sOff = (int*)(sX + 2*RPB);                            // 64 i32 (H base offsets)

    const int tid   = threadIdx.x;
    const int w     = tid >> 5;          // wave 0..7 -> owns gh columns 16w..16w+15 of each gate
    const int l     = tid & 31;
    const int col16 = l & 15;
    const int hh8   = (l >> 4) << 3;     // 0 or 8 (half-wave K/M offset)
    const int j     = w*16 + col16;      // hidden column 0..127 owned by this lane
    const int r0    = blockIdx.x * RPB;  // first chain row of this block

    // ---- one-time staging: W_hh -> f16 LDS; zero initial h, x; per-row H offsets ----
    for (int i = tid; i < G3*HID; i += 256) sWhh[i] = (_Float16)Whh[i];
    for (int i = tid; i < RPB*HID; i += 256) sH[i] = (_Float16)0.f;
    if (tid < RPB) {
        sX[tid] = 0.f;
        const int row = r0 + tid;
        const int b = row / NN, n = row - b*NN;
        sOff[tid] = (b*TT*NN + n)*HID;                 // + j + t*N*HID at store time
    }
    __syncthreads();

    // ---- per-lane gate constants ----
    float wx0[3], wx1[3], bi2, bsum[2];
#pragma unroll
    for (int g = 0; g < 3; ++g) {
        const int c = g*HID + j;
        wx0[g] = Wih[2*c]; wx1[g] = Wih[2*c + 1];
        if (g < 2) bsum[g] = bih[c] + bhh[c];
    }
    bi2 = bih[2*HID + j];
    const float bhhn = bhh[2*HID + j];
    const float fw = fcw[j];
    const float fb = fcb[0];

    // per-lane B-fragment base inside sWhh: column c = g*128 + j, row-major stride HID
    const _Float16* bbase = sWhh + j*HID + hh8;        // + g*HID*HID + kk*32

    float hOld[4][8] = {};                             // z*h_old carry, fp32, in regs

    // ---- time loop (sequential recurrence) ----
#pragma unroll 1
    for (int t = 0; t < TT; ++t) {
        const int cur = t & 1, nxt = cur ^ 1;
        if (tid < RPB) {
            const int row = r0 + tid;
            const int b = row / NN, n = row - b*NN;
            sPm[tid] = pm25[(b*TT + t)*NN + n];
            sX[nxt*RPB + tid] = fb;                    // seed x accumulator with fc_b
        }
        __syncthreads();

        const _Float16* hcur = sH + cur*RPB*HID;
        _Float16*       hnxt = sH + nxt*RPB*HID;
        const float*    xc   = sX + cur*RPB;
        float*          xn   = sX + nxt*RPB;
        const int tOff = t * NN * HID + j;

#pragma unroll
        for (int mt = 0; mt < 4; ++mt) {
            const int mbase = mt*16 + hh8;
            const int arow  = mt*16 + col16;

            // seed accumulators and pre-fold i_n; xv/pv live only in this scope
            v8f accR, accZ, accN;
            float inN[8];
            {
                v4f xlo = *(const v4f*)&xc[mbase],  xhi = *(const v4f*)&xc[mbase+4];
                v4f plo = *(const v4f*)&sPm[mbase], phi = *(const v4f*)&sPm[mbase+4];
#pragma unroll
                for (int v = 0; v < 8; ++v) {
                    const float xv = (v < 4) ? xlo[v & 3] : xhi[v & 3];
                    const float pv = (v < 4) ? plo[v & 3] : phi[v & 3];
                    accR[v] = wx0[0]*xv + wx1[0]*pv + bsum[0];   // gi_r + b_ih_r + b_hh_r
                    accZ[v] = wx0[1]*xv + wx1[1]*pv + bsum[1];   // gi_z + b_ih_z + b_hh_z
                    accN[v] = bhhn;                               // only b_hh_n (r scales gh_n+b_hh_n)
                    inN[v]  = wx0[2]*xv + wx1[2]*pv + bi2;        // i_n kept separate
                }
            }

            // gh = h @ W_hh^T : K = 128 in 4 chunks of 32.
            // A and B fragments streamed from LDS (keeps VGPR count < 256).
#pragma unroll
            for (int kk = 0; kk < 4; ++kk) {
                const v16h A  = load_frag(&hcur[arow*HID + kk*32 + hh8]);
                const v16h B0 = load_frag(bbase + 0*HID*HID + kk*32);
                const v16h B1 = load_frag(bbase + 1*HID*HID + kk*32);
                const v16h B2 = load_frag(bbase + 2*HID*HID + kk*32);
                accR = __builtin_amdgcn_wmma_f32_16x16x32_f16(false, A, false, B0, (short)0, accR, false, false);
                accZ = __builtin_amdgcn_wmma_f32_16x16x32_f16(false, A, false, B1, (short)0, accZ, false, false);
                accN = __builtin_amdgcn_wmma_f32_16x16x32_f16(false, A, false, B2, (short)0, accN, false, false);
            }

            // per-row H output base offsets for this tile (two b128 LDS reads)
            v4i oLo = *(const v4i*)&sOff[mbase];
            v4i oHi = *(const v4i*)&sOff[mbase + 4];

            // gates + state update + outputs (all in registers; acc layout => matching triples)
#pragma unroll
            for (int v = 0; v < 8; ++v) {
                const float r  = sigf(accR[v]);
                const float z  = sigf(accZ[v]);
                const float nt = tanhf_(inN[v] + r*accN[v]);
                const float hn = (1.0f - z)*nt + z*hOld[mt][v];
                hOld[mt][v] = hn;

                const int off = ((v < 4) ? oLo[v & 3] : oHi[v & 3]) + tOff;
                out[off] = hn;                                      // stream H (coalesced per half-wave)
                hnxt[(mbase + v)*HID + j] = (_Float16)hn;           // f16 h for next step's A frags

                // x_new: width-16 shuffle-tree dot with fc_w, then one ds_add per row per wave
                float p = hn * fw;
                p += __shfl_xor(p, 1, 16);
                p += __shfl_xor(p, 2, 16);
                p += __shfl_xor(p, 4, 16);
                p += __shfl_xor(p, 8, 16);
                if (col16 == 0) atomicAdd(&xn[mbase + v], p);
            }
        }
        __syncthreads();   // h_f16[nxt] and x[nxt] complete before next step

        if (t == TT-1 && tid < RPB)
            out[HSIZE + r0 + tid] = sX[nxt*RPB + tid];              // xn_final
    }
}

extern "C" void kernel_launch(void* const* d_in, const int* in_sizes, int n_in,
                              void* d_out, int out_size, void* d_ws, size_t ws_size,
                              hipStream_t stream) {
    const float* pm25 = (const float*)d_in[0];
    const float* Wih  = (const float*)d_in[1];
    const float* Whh  = (const float*)d_in[2];
    const float* bih  = (const float*)d_in[3];
    const float* bhh  = (const float*)d_in[4];
    const float* fcw  = (const float*)d_in[5];
    const float* fcb  = (const float*)d_in[6];
    float* out = (float*)d_out;

    const size_t shmem = (size_t)G3*HID*2      // W_hh f16
                       + (size_t)2*RPB*HID*2   // h ping-pong f16
                       + (size_t)RPB*4         // pm
                       + (size_t)2*RPB*4       // x ping-pong
                       + (size_t)RPB*4;        // H base offsets
    const int blocks = (BB*NN) / RPB;          // 184
    gru_encoder_wmma<<<blocks, 256, shmem, stream>>>(pm25, Wih, Whh, bih, bhh, fcw, fcb, out);
}
